// CK3DAttention_10307921510567
// MI455X (gfx1250) — matrix operand order repaired and checked by reading, not compiled, not run on hardware
//
#include <hip/hip_runtime.h>
#include <hip/hip_bf16.h>
#include <math.h>

// ---------------------------------------------------------------------------
// CDNA5 (gfx1250, wave32) implementation of the gated axial-attention block.
// All dense math goes through v_wmma_f32_16x16x32_bf16 (f32 accumulate).
// GEMM X tiles are staged with gfx1250 async global->LDS copies (ASYNCcnt);
// LDS fragment loads are packed b128.
// ---------------------------------------------------------------------------

typedef __attribute__((ext_vector_type(16))) __bf16 v16bf;
typedef __attribute__((ext_vector_type(8)))  __bf16 v8bf;
typedef __attribute__((ext_vector_type(8)))  float  v8f;
typedef __attribute__((ext_vector_type(4)))  int    v4i;

#define GS 32768   // T*H*W spatial extent per (b, channel)
#define CIN 256

#if defined(__has_builtin)
#  if __has_builtin(__builtin_amdgcn_global_load_async_to_lds_b128) && \
      __has_builtin(__builtin_amdgcn_s_wait_asynccnt)
#    define USE_ASYNC 1
#  endif
#endif
#ifndef USE_ASYNC
#  define USE_ASYNC 0
#endif

#if USE_ASYNC
typedef __attribute__((address_space(1))) v4i* gptr_b128;   // global int4*
typedef __attribute__((address_space(3))) v4i* lptr_b128;   // LDS int4*
#endif

// Load 16 consecutive bf16 (32B) from LDS as two 16B chunks -> v16bf fragment.
__device__ __forceinline__ v16bf lds_ld16(const __bf16* p) {
  v8bf a = *(const v8bf*)p;
  v8bf b = *(const v8bf*)(p + 8);
  return __builtin_shufflevector(a, b, 0,1,2,3,4,5,6,7,8,9,10,11,12,13,14,15);
}

// Store 16 bf16 to LDS as two b128 stores.
__device__ __forceinline__ void lds_st16(__bf16* p, v16bf v) {
  *(v8bf*)p       = __builtin_shufflevector(v, v, 0,1,2,3,4,5,6,7);
  *(v8bf*)(p + 8) = __builtin_shufflevector(v, v, 8,9,10,11,12,13,14,15);
}

__device__ __forceinline__ v8bf pack8(const float* f) {
  v8bf r;
  #pragma unroll
  for (int j = 0; j < 8; ++j) r[j] = (__bf16)f[j];
  return r;
}

__device__ __forceinline__ v8f wmma_bf16(v16bf a, v16bf b, v8f c) {
  return __builtin_amdgcn_wmma_f32_16x16x32_bf16(false, a, false, b, (short)0, c,
                                                 false, false);
}

// ---------------------------------------------------------------------------
// GEMM: Y[b][o][s] = sum_c W[o][c] * X[b][c][s]    (K = 256, bf16 WMMA)
// Block tile 64(o) x 128(s), 8 waves: wave = (ow 0..3, sw 0..1); each wave
// owns 16(o) x 64(s) = four 16x16 accumulators; K stepped by 32.
// X tiles: raw f32 tile double-buffered in LDS via async global->LDS b128
// copies issued one K-step ahead (overlaps HBM latency with WMMA);
// then converted in LDS to the bf16 fragment swizzle:
//   A (16-bit 16x32): lane l: row=l&15, half h=l>>4; e <-> K = 16h + e
//   B (16-bit 32x16): lane l: col=l&15, group g=l>>4; e <-> K = 16g + e
// Rows padded to 24 elems (48B) for bank spread, 16B aligned.
// ---------------------------------------------------------------------------
__global__ __launch_bounds__(256) void gemm_wmma(
    const float* __restrict__ Wm,   // O x 256
    const float* __restrict__ X,    // B x 256 x GS
    float* __restrict__ Y,          // B x O x GS
    int O)
{
  __shared__ __align__(16) __bf16 lw[2][64][24];
  __shared__ __align__(16) __bf16 lx[2][128][24];

  const int tid  = threadIdx.x;
  const int lane = tid & 31;
  const int wv   = tid >> 5;
  const int ow   = wv >> 1;        // o sub-tile 0..3
  const int sw   = wv & 1;         // s sub-tile 0..1
  const int s0   = blockIdx.x * 128;
  const int o0   = blockIdx.y * 64;
  const int b    = blockIdx.z;

  const float* Xb = X + (size_t)b * CIN * GS;
  float*       Yb = Y + (size_t)b * O * GS;

  v8f acc[4] = {};

  // W-tile staging map: thread -> (o row, 8-wide k group).
  const int wol = tid >> 2, wkg = tid & 3;

#if USE_ASYNC
  __shared__ __align__(16) float lraw[2][32][128];   // raw f32 X tiles
  auto async_x = [&](int kkn, int buf) {
    #pragma unroll
    for (int r = 0; r < 4; ++r) {
      int ch = tid + 256 * r;          // 1024 chunks of 16B (32k x 128s f32)
      int kr = ch >> 5, cs = ch & 31;
      const float* src = &Xb[(size_t)(kkn * 32 + kr) * GS + s0 + cs * 4];
      __builtin_amdgcn_global_load_async_to_lds_b128(
          (gptr_b128)(void*)src,
          (lptr_b128)(void*)&lraw[buf][kr][cs * 4],
          0, 0);
    }
  };
  async_x(0, 0);
#endif

  for (int kk = 0; kk < 8; ++kk) {
#if USE_ASYNC
    const int p = kk & 1;
    if (kk < 7) {
      async_x(kk + 1, 1 - p);                  // prefetch next tile via DMA
      __builtin_amdgcn_s_wait_asynccnt(4);     // previous-tile copies done
    } else {
      __builtin_amdgcn_s_wait_asynccnt(0);
    }
#endif
    __syncthreads();   // tile visible to all waves; lx reads of prior step done

    // Stage W tile (64 o x 32 k): one 32B global read + one b128 LDS store.
    {
      const float* src = &Wm[(size_t)(o0 + wol) * CIN + kk * 32 + wkg * 8];
      float4 a = *(const float4*)src;
      float4 c = *(const float4*)(src + 4);
      float f[8] = {a.x, a.y, a.z, a.w, c.x, c.y, c.z, c.w};
      *(v8bf*)&lw[wkg >> 1][wol][(wkg & 1) * 8] = pack8(f);
    }

#if USE_ASYNC
    // Convert raw f32 tile -> bf16 B-fragment swizzle.
    #pragma unroll
    for (int r = 0; r < 2; ++r) {
      int uu = tid + 256 * r;
      int sl = uu & 127, kg = uu >> 7;
      float f[8];
      #pragma unroll
      for (int j = 0; j < 8; ++j) f[j] = lraw[p][kg * 8 + j][sl];
      *(v8bf*)&lx[kg >> 1][sl][(kg & 1) * 8] = pack8(f);
    }
#else
    // Fallback: direct global staging (8 coalesced b32 reads -> b128 store).
    #pragma unroll
    for (int r = 0; r < 2; ++r) {
      int uu = tid + 256 * r;
      int sl = uu & 127, kg = uu >> 7;
      float f[8];
      #pragma unroll
      for (int j = 0; j < 8; ++j)
        f[j] = Xb[(size_t)(kk * 32 + kg * 8 + j) * GS + s0 + sl];
      *(v8bf*)&lx[kg >> 1][sl][(kg & 1) * 8] = pack8(f);
    }
    if (kk < 7)
      __builtin_prefetch(&Xb[(size_t)((kk + 1) * 32 + (tid >> 7) * 8) * GS + s0 + (tid & 127)], 0, 3);
#endif
    __syncthreads();

    v16bf a = lds_ld16(&lw[lane >> 4][ow * 16 + (lane & 15)][0]);
    #pragma unroll
    for (int t = 0; t < 4; ++t) {
      v16bf bt = lds_ld16(&lx[lane >> 4][sw * 64 + t * 16 + (lane & 15)][0]);
      acc[t] = wmma_bf16(a, bt, acc[t]);
    }
  }

  // D fragment: lane l: col n=l&15, rows m = 8*(l>>4)+i.
  const int n = lane & 15, hh = lane >> 4;
  #pragma unroll
  for (int t = 0; t < 4; ++t) {
    #pragma unroll
    for (int i = 0; i < 8; ++i) {
      int m = o0 + ow * 16 + hh * 8 + i;
      Yb[(size_t)m * GS + s0 + sw * 64 + t * 16 + n] = acc[t][i];
    }
  }
}

// ---------------------------------------------------------------------------
// qg[b][c][t] = mean_{h,w} q[b][c][t][h][w]   (q = channels [0,256) of qkv)
// ---------------------------------------------------------------------------
__global__ __launch_bounds__(256) void reduce_qg(const float* __restrict__ QKV,
                                                 float* __restrict__ QG) {
  int id = blockIdx.x;                       // b*8192 + c*32 + t
  int b = id >> 13, c = (id >> 5) & 255, t = id & 31;
  const float* p = QKV + (size_t)b * 768 * GS + (size_t)c * GS + (size_t)t * 1024;
  float s = 0.f;
  for (int i = threadIdx.x; i < 1024; i += 256) s += p[i];
  __shared__ float red[256];
  red[threadIdx.x] = s;
  __syncthreads();
  for (int w = 128; w > 0; w >>= 1) {
    if (threadIdx.x < w) red[threadIdx.x] += red[threadIdx.x + w];
    __syncthreads();
  }
  if (threadIdx.x == 0) QG[id] = red[0] * (1.0f / 1024.0f);
}

// h[b][o][t] = gelu(pre_w @ qg + pre_b)  (exact erf gelu)
__global__ __launch_bounds__(256) void mix_pre_k(const float* __restrict__ QG,
                                                 const float* __restrict__ PW,
                                                 const float* __restrict__ PB,
                                                 float* __restrict__ Hb) {
  int id = blockIdx.x * 256 + threadIdx.x;   // 16384
  int b = id >> 13, o = (id >> 5) & 255, t = id & 31;
  float s = PB[o];
  const float* qg = QG + b * 8192 + t;       // [b][c][t], c stride 32
  for (int c = 0; c < 256; ++c) s += PW[o * 256 + c] * qg[c * 32];
  Hb[id] = 0.5f * s * (1.0f + erff(s * 0.70710678f));
}

// alpha[b][m][t] = softmax_m( causal conv1d(h, cw)[m][t] + cb[m] )
__global__ void mix_alpha_k(const float* __restrict__ Hb,
                            const float* __restrict__ CW,
                            const float* __restrict__ CB,
                            float* __restrict__ AL) {
  int id = blockIdx.x * blockDim.x + threadIdx.x;   // B*T = 64
  if (id >= 64) return;
  int b = id >> 5, t = id & 31;
  float lg[3];
  for (int m = 0; m < 3; ++m) {
    float s = CB[m];
    for (int c = 0; c < 256; ++c) {
      const float* w  = CW + (m * 256 + c) * 3;
      const float* hh = Hb + b * 8192 + c * 32;
      #pragma unroll
      for (int j = 0; j < 3; ++j) {
        int ti = t + j - 2;
        if (ti >= 0) s += w[j] * hh[ti];
      }
    }
    lg[m] = s;
  }
  float mx = fmaxf(lg[0], fmaxf(lg[1], lg[2]));
  float e0 = __expf(lg[0] - mx), e1 = __expf(lg[1] - mx), e2 = __expf(lg[2] - mx);
  float r = 1.f / (e0 + e1 + e2);
  AL[(b * 3 + 0) * 32 + t] = e0 * r;
  AL[(b * 3 + 1) * 32 + t] = e1 * r;
  AL[(b * 3 + 2) * 32 + t] = e2 * r;
}

// ---------------------------------------------------------------------------
// k_out/v_out = sum_m alpha[b][m][t] * depthwise_causal_conv3d(k/v, bank[m]).
// (b, c, t) is uniform within a block (block covers 8 h-rows x 32 w), so the
// 54 alpha-mixed effective taps are computed once into LDS.
// ---------------------------------------------------------------------------
__global__ __launch_bounds__(256) void conv_mix_k(
    const float* __restrict__ QKV, const float* __restrict__ WK,
    const float* __restrict__ WV, const float* __restrict__ AL,
    float* __restrict__ KO, float* __restrict__ VO)
{
  __shared__ float swk[27], swv[27];
  size_t id = (size_t)blockIdx.x * 256 + threadIdx.x;   // 16,777,216
  int w   = (int)(id & 31);
  int hwi = (int)((id >> 5) & 31);
  int t   = (int)((id >> 10) & 31);   // uniform in block
  int c   = (int)((id >> 15) & 255);  // uniform in block
  int b   = (int)(id >> 23);          // uniform in block

  if (threadIdx.x < 54) {
    int tap = threadIdx.x < 27 ? (int)threadIdx.x : (int)threadIdx.x - 27;
    const float* bank = threadIdx.x < 27 ? WK : WV;
    float* dst        = threadIdx.x < 27 ? swk : swv;
    float a0 = AL[(b * 3 + 0) * 32 + t];
    float a1 = AL[(b * 3 + 1) * 32 + t];
    float a2 = AL[(b * 3 + 2) * 32 + t];
    dst[tap] = a0 * bank[(size_t)c * 27 + tap]
             + a1 * bank[(size_t)(256 + c) * 27 + tap]
             + a2 * bank[(size_t)(512 + c) * 27 + tap];
  }
  __syncthreads();

  const float* kin = QKV + (size_t)b * 768 * GS + (size_t)(256 + c) * GS;
  const float* vin = QKV + (size_t)b * 768 * GS + (size_t)(512 + c) * GS;
  float sk = 0.f, sv = 0.f;
  for (int dt = 0; dt < 3; ++dt) {
    int ti = t + dt - 2;
    if (ti < 0) continue;
    for (int dh = 0; dh < 3; ++dh) {
      int hi = hwi + dh - 1;
      if ((unsigned)hi >= 32u) continue;
      #pragma unroll
      for (int dw = 0; dw < 3; ++dw) {
        int wi = w + dw - 1;
        if ((unsigned)wi >= 32u) continue;
        int tap = (dt * 3 + dh) * 3 + dw;
        float xk = kin[(size_t)ti * 1024 + hi * 32 + wi];
        float xv = vin[(size_t)ti * 1024 + hi * 32 + wi];
        sk += swk[tap] * xk;
        sv += swv[tap] * xv;
      }
    }
  }
  size_t o = (size_t)b * 256 * GS + (size_t)c * GS + (size_t)t * 1024 + hwi * 32 + w;
  KO[o] = sk;
  VO[o] = sv;
}

// ---------------------------------------------------------------------------
// Axial attention. One WAVE owns one (b, head, line) problem: L=32, d=32.
//   S^T = K_rope (32x32, K-dim=d) * Q_rope^T   -> softmax over k (rows of S^T)
//   ctx^T = V^T (d x k) * P^T (k x q)
// Fragment staging (RoPE applied in-flight, lane = sequence position):
//   Q -> B-frag layout  lq[g=dd>>4][pos][e=dd&15]      (packed b128 stores)
//   K -> A-frag layout  lk[h(dd)][pos][e(dd)]          (packed b128 stores)
//   V -> A-frag layout over k: lv[h(pos)][dd][e(pos)]  (true transpose, scalar)
// ---------------------------------------------------------------------------
__global__ __launch_bounds__(128) void axial_attn(
    const float* __restrict__ QKV, const float* __restrict__ K,
    const float* __restrict__ V, float* __restrict__ CTX, int axis)
{
  __shared__ __align__(16) __bf16 lq[4][2][32][24];
  __shared__ __align__(16) __bf16 lk[4][2][32][24];
  __shared__ __align__(16) __bf16 lv[4][2][32][24];

  const int lane = threadIdx.x & 31;
  const int wv   = threadIdx.x >> 5;
  const unsigned u = blockIdx.x * 4 + wv;       // 0..16383
  const int b    = u >> 13;
  const int head = (u >> 10) & 7;
  const int sp   = u & 1023;

  int unitoff, stride;
  if (axis == 0)      { unitoff = sp;                              stride = 1024; } // T
  else if (axis == 1) { unitoff = (sp >> 5) * 1024 + (sp & 31);    stride = 32;   } // H
  else                { unitoff = sp * 32;                         stride = 1;    } // W
  const bool causal = (axis == 0);

  const size_t qbase = (size_t)b * 768 * GS + (size_t)head * 32 * GS + unitoff;
  const size_t kbase = (size_t)b * 256 * GS + (size_t)head * 32 * GS + unitoff;

  const int pos = lane;
  const size_t poff = (size_t)pos * stride;
  const int hk = pos >> 4;
  const int ek = pos & 15;

  // Stage with RoPE (rotate-half pair kept in-lane: dd and dd+16).
  v16bf qr0, qr1, kr0, kr1;
  #pragma unroll
  for (int dd = 0; dd < 16; ++dd) {
    float fr = __expf(-(float)dd * 0.57564627f);  // 1/10000^(dd/16)
    float cs, sn;
    __sincosf((float)pos * fr, &cs, &sn);
    float qa = QKV[qbase + (size_t)dd * GS + poff];
    float qb = QKV[qbase + (size_t)(dd + 16) * GS + poff];
    float ka = K[kbase + (size_t)dd * GS + poff];
    float kb = K[kbase + (size_t)(dd + 16) * GS + poff];
    float va = V[kbase + (size_t)dd * GS + poff];
    float vb = V[kbase + (size_t)(dd + 16) * GS + poff];
    float q0 = qa * cs - qb * sn, q1 = qb * cs + qa * sn;
    float k0 = ka * cs - kb * sn, k1 = kb * cs + ka * sn;
    qr0[dd] = (__bf16)q0;
    qr1[dd] = (__bf16)q1;
    kr0[dd] = (__bf16)k0;
    kr1[dd] = (__bf16)k1;
    lv[wv][hk][dd][ek]      = (__bf16)va;
    lv[wv][hk][dd + 16][ek] = (__bf16)vb;
  }
  lds_st16(&lq[wv][0][pos][0], qr0);
  lds_st16(&lq[wv][1][pos][0], qr1);
  lds_st16(&lk[wv][0][pos][0], kr0);
  lds_st16(&lk[wv][1][pos][0], kr1);

  const int nlo = lane & 15, hh = lane >> 4;
  v16bf aK0 = lds_ld16(&lk[wv][hh][nlo][0]);        // k rows 0..15
  v16bf aK1 = lds_ld16(&lk[wv][hh][16 + nlo][0]);   // k rows 16..31
  v16bf bQ0 = lds_ld16(&lq[wv][hh][nlo][0]);        // q cols 0..15
  v16bf bQ1 = lds_ld16(&lq[wv][hh][16 + nlo][0]);   // q cols 16..31

  v8f z = {};
  v8f s00 = wmma_bf16(aK0, bQ0, z);
  v8f s10 = wmma_bf16(aK1, bQ0, z);
  v8f s01 = wmma_bf16(aK0, bQ1, z);
  v8f s11 = wmma_bf16(aK1, bQ1, z);

  const float sc = 0.17677669529663687f;           // 1/sqrt(32)
  const int q0c = nlo, q1c = 16 + nlo;
  #pragma unroll
  for (int i = 0; i < 8; ++i) {
    int k0r = hh * 8 + i, k1r = 16 + hh * 8 + i;
    s00[i] = (causal && k0r > q0c) ? -1e30f : s00[i] * sc;
    s10[i] = (causal && k1r > q0c) ? -1e30f : s10[i] * sc;
    s01[i] = (causal && k0r > q1c) ? -1e30f : s01[i] * sc;
    s11[i] = (causal && k1r > q1c) ? -1e30f : s11[i] * sc;
  }

  // Softmax over k (per q column; lanes l and l+16 share the same q).
  float m0 = -1e30f, m1 = -1e30f;
  #pragma unroll
  for (int i = 0; i < 8; ++i) {
    m0 = fmaxf(m0, fmaxf(s00[i], s10[i]));
    m1 = fmaxf(m1, fmaxf(s01[i], s11[i]));
  }
  m0 = fmaxf(m0, __shfl_xor(m0, 16, 32));
  m1 = fmaxf(m1, __shfl_xor(m1, 16, 32));
  float sum0 = 0.f, sum1 = 0.f;
  #pragma unroll
  for (int i = 0; i < 8; ++i) {
    s00[i] = __expf(s00[i] - m0); s10[i] = __expf(s10[i] - m0);
    s01[i] = __expf(s01[i] - m1); s11[i] = __expf(s11[i] - m1);
    sum0 += s00[i] + s10[i];
    sum1 += s01[i] + s11[i];
  }
  sum0 += __shfl_xor(sum0, 16, 32);
  sum1 += __shfl_xor(sum1, 16, 32);
  float r0 = 1.f / sum0, r1 = 1.f / sum1;
  #pragma unroll
  for (int i = 0; i < 8; ++i) {
    s00[i] *= r0; s10[i] *= r0; s01[i] *= r1; s11[i] *= r1;
  }

  // Repack P (D-frag of S^T, f32) into bf16 B-fragments for ctx^T = V^T * P.
  // Half-wave exchange: each lane sends the frag its partner needs.
  v16bf bP0, bP1;
  #pragma unroll
  for (int i = 0; i < 8; ++i) {
    float t0 = __shfl_xor(hh ? s00[i] : s10[i], 16, 32);
    float t1 = __shfl_xor(hh ? s01[i] : s11[i], 16, 32);
    bP0[i]     = (__bf16)(hh ? t0 : s00[i]);
    bP0[i + 8] = (__bf16)(hh ? s10[i] : t0);
    bP1[i]     = (__bf16)(hh ? t1 : s01[i]);
    bP1[i + 8] = (__bf16)(hh ? s11[i] : t1);
  }

  v16bf aV0 = lds_ld16(&lv[wv][hh][nlo][0]);        // d rows 0..15
  v16bf aV1 = lds_ld16(&lv[wv][hh][16 + nlo][0]);   // d rows 16..31
  v8f c00 = wmma_bf16(aV0, bP0, z);
  v8f c10 = wmma_bf16(aV1, bP0, z);
  v8f c01 = wmma_bf16(aV0, bP1, z);
  v8f c11 = wmma_bf16(aV1, bP1, z);

  const size_t obase = (size_t)b * 256 * GS + (size_t)head * 32 * GS + unitoff;
  #pragma unroll
  for (int i = 0; i < 8; ++i) {
    int d0 = hh * 8 + i;
    CTX[obase + (size_t)d0 * GS + (size_t)q0c * stride]        = c00[i];
    CTX[obase + (size_t)(16 + d0) * GS + (size_t)q0c * stride] = c10[i];
    CTX[obase + (size_t)d0 * GS + (size_t)q1c * stride]        = c01[i];
    CTX[obase + (size_t)(16 + d0) * GS + (size_t)q1c * stride] = c11[i];
  }
}

// y = sigmoid(g_t)*ctx_t + sigmoid(g_h)*ctx_h + sigmoid(g_w)*ctx_w
__global__ __launch_bounds__(256) void combine_k(
    const float* __restrict__ G, const float* __restrict__ GB,
    const float* __restrict__ CT, const float* __restrict__ CH,
    const float* __restrict__ CWb, float* __restrict__ Y)
{
  size_t id = (size_t)blockIdx.x * 256 + threadIdx.x;   // 16,777,216
  int s = (int)(id & 32767);
  int c = (int)((id >> 15) & 255);
  int b = (int)(id >> 23);
  size_t gbase = (size_t)b * 768 * GS + s;
  float gt = G[gbase + (size_t)c * GS] + GB[c];
  float gh = G[gbase + (size_t)(256 + c) * GS] + GB[256 + c];
  float gw = G[gbase + (size_t)(512 + c) * GS] + GB[512 + c];
  gt = 1.f / (1.f + __expf(-gt));
  gh = 1.f / (1.f + __expf(-gh));
  gw = 1.f / (1.f + __expf(-gw));
  size_t ci = (size_t)b * 256 * GS + (size_t)c * GS + s;
  Y[ci] = gt * CT[ci] + gh * CH[ci] + gw * CWb[ci];
}

// ---------------------------------------------------------------------------
extern "C" void kernel_launch(void* const* d_in, const int* in_sizes, int n_in,
                              void* d_out, int out_size, void* d_ws, size_t ws_size,
                              hipStream_t stream) {
  const float* x      = (const float*)d_in[0];
  const float* w_qkv  = (const float*)d_in[1];
  const float* wk     = (const float*)d_in[2];
  const float* wvb    = (const float*)d_in[3];
  const float* pre_w  = (const float*)d_in[4];
  const float* pre_b  = (const float*)d_in[5];
  const float* cw     = (const float*)d_in[6];
  const float* cb     = (const float*)d_in[7];
  const float* gate_w = (const float*)d_in[8];
  const float* gate_b = (const float*)d_in[9];
  const float* proj_w = (const float*)d_in[10];
  float* out = (float*)d_out;

  float* ws   = (float*)d_ws;
  float* qkv  = ws;                      // 2*768*GS
  float* g    = qkv  + (size_t)50331648; // 2*768*GS
  float* ko   = g    + (size_t)50331648; // 2*256*GS
  float* vo   = ko   + (size_t)16777216;
  float* ctxT = vo   + (size_t)16777216;
  float* ctxH = ctxT + (size_t)16777216;
  float* ctxW = ctxH + (size_t)16777216;
  float* yb   = ctxW + (size_t)16777216;
  float* qg   = yb   + (size_t)16777216; // 2*256*32
  float* hb   = qg   + (size_t)16384;
  float* al   = hb   + (size_t)16384;    // 2*3*32

  gemm_wmma<<<dim3(256, 12, 2), 256, 0, stream>>>(w_qkv,  x,  qkv, 768);
  gemm_wmma<<<dim3(256, 12, 2), 256, 0, stream>>>(gate_w, x,  g,   768);
  reduce_qg<<<16384, 256, 0, stream>>>(qkv, qg);
  mix_pre_k<<<64, 256, 0, stream>>>(qg, pre_w, pre_b, hb);
  mix_alpha_k<<<1, 64, 0, stream>>>(hb, cw, cb, al);
  conv_mix_k<<<65536, 256, 0, stream>>>(qkv, wk, wvb, al, ko, vo);
  axial_attn<<<4096, 128, 0, stream>>>(qkv, ko, vo, ctxT, 0);
  axial_attn<<<4096, 128, 0, stream>>>(qkv, ko, vo, ctxH, 1);
  axial_attn<<<4096, 128, 0, stream>>>(qkv, ko, vo, ctxW, 2);
  combine_k<<<65536, 256, 0, stream>>>(g, gate_b, ctxT, ctxH, ctxW, yb);
  gemm_wmma<<<dim3(256, 4, 2), 256, 0, stream>>>(proj_w, yb, out, 256);
}
